// MultiHeadAttention_14645838479733
// MI455X (gfx1250) — compile-verified, matrix operand order checked by reading, not run on hardware
//
#include <hip/hip_runtime.h>
#include <stdint.h>

// MI455X / gfx1250, wave32. WMMA f32_16x16x32_f16 + async LDS staging.

typedef _Float16 half_t;
typedef __attribute__((ext_vector_type(16))) _Float16 v16h;
typedef __attribute__((ext_vector_type(8)))  _Float16 v8h;
typedef __attribute__((ext_vector_type(8)))  float    v8f;
typedef __attribute__((ext_vector_type(4)))  float    v4f;

#define BATCH  2
#define SEQ    2048
#define DMODEL 1024
#define NHEADS 16
#define HDIM   64

// ---- WMMA operand packing ----------------------------------------------
// A-layout (16x32 f16): lane L holds row M=L%16; K halves {g0..g0+7} and
// {g0+16..g0+23}, g0 = (L<16?0:8).  B-layout (32x16): lane=col, same groups.
static __device__ inline v16h pack_f16(const half_t* __restrict__ p, int g0) {
  v8h a = *(const v8h*)(p + g0);
  v8h b = *(const v8h*)(p + g0 + 16);
  v16h r;
#pragma unroll
  for (int i = 0; i < 8; ++i) { r[i] = a[i]; r[8 + i] = b[i]; }
  return r;
}

static __device__ inline v8f wmma_f16(v16h a, v16h b, v8f c) {
  return __builtin_amdgcn_wmma_f32_16x16x32_f16(false, a, false, b,
                                                (short)0, c, false, false);
}

// ---- CDNA5 async global->LDS copy (one 16B chunk per lane) -------------
static __device__ inline void async_ld16(unsigned lds_off,
                                         unsigned long long base,
                                         unsigned goff) {
  asm volatile("global_load_async_to_lds_b128 %0, %1, %2 offset:0"
               :: "v"(lds_off), "v"(goff), "s"(base)
               : "memory");
}
static __device__ inline void wait_async_le2() {
  asm volatile("s_wait_asynccnt 0x2" ::: "memory");
}

// ---- Kernel 0: fp32 -> f16 elementwise (amortizes all GEMM converts) ---
__global__ __launch_bounds__(256) void cvt_f16_kernel(
    const float* __restrict__ src, half_t* __restrict__ dst, int n8) {
  const int i = blockIdx.x * blockDim.x + threadIdx.x;
  if (i < n8) {
    const v4f* s = (const v4f*)(src + (size_t)i * 8);
    v4f a = s[0], b = s[1];
    v8h r;
#pragma unroll
    for (int j = 0; j < 4; ++j) {
      r[j]     = (_Float16)a[j];
      r[4 + j] = (_Float16)b[j];
    }
    *(v8h*)(dst + (size_t)i * 8) = r;
  }
}

// ---- Kernel 1: projection GEMM  y = x @ W^T  (f16 in, f16 out) ---------
// 2x2 register-blocked: each wave computes a 32x32 output block (4 WMMA
// accumulators), halving operand loads per WMMA vs single-tile.
// mode 0: Q -> [B,H,S,Dh] scaled by 1/sqrt(Dh); 1: K -> [B,H,S,Dh];
// mode 2: V -> transposed [B,H,Dh,S]
__global__ __launch_bounds__(256) void mha_proj_kernel(
    const half_t* __restrict__ x, const half_t* __restrict__ w,
    half_t* __restrict__ out, int mode) {
  const int lane   = threadIdx.x & 31;
  const int wave   = (blockIdx.x * blockDim.x + threadIdx.x) >> 5;
  const int TN2    = DMODEL / 32;                 // 32 col-blocks
  const int tn2    = wave & (TN2 - 1);
  const int tm2    = wave / TN2;                  // 0..127
  const int half16 = lane >> 4;
  const int g0     = half16 << 3;
  const int rloc   = lane & 15;

  const half_t* xrow0 = x + (size_t)(tm2 * 32 + rloc) * DMODEL;
  const half_t* xrow1 = xrow0 + (size_t)16 * DMODEL;
  const half_t* wrow0 = w + (size_t)(tn2 * 32 + rloc) * DMODEL;
  const half_t* wrow1 = wrow0 + (size_t)16 * DMODEL;

  v8f acc[2][2] = {{{}, {}}, {{}, {}}};
#pragma unroll 2
  for (int kk = 0; kk < DMODEL; kk += 32) {
    const v16h a0 = pack_f16(xrow0 + kk, g0);
    const v16h a1 = pack_f16(xrow1 + kk, g0);
    const v16h b0 = pack_f16(wrow0 + kk, g0);
    const v16h b1 = pack_f16(wrow1 + kk, g0);
    acc[0][0] = wmma_f16(a0, b0, acc[0][0]);
    acc[0][1] = wmma_f16(a0, b1, acc[0][1]);
    acc[1][0] = wmma_f16(a1, b0, acc[1][0]);
    acc[1][1] = wmma_f16(a1, b1, acc[1][1]);
  }

#pragma unroll
  for (int mi = 0; mi < 2; ++mi)
#pragma unroll
    for (int ni = 0; ni < 2; ++ni) {
      const int col   = tn2 * 32 + ni * 16 + rloc;
      const int h     = col >> 6;
      const int dh    = col & 63;
      const int mrow0 = tm2 * 32 + mi * 16 + (half16 << 3);
      const int b     = mrow0 / SEQ;
      const int s0    = mrow0 & (SEQ - 1);
      if (mode == 2) {             // V transposed: 8 consecutive seq slots
        v8h pk;
#pragma unroll
        for (int j = 0; j < 8; ++j) pk[j] = (_Float16)acc[mi][ni][j];
        half_t* dst =
            out + ((size_t)(b * NHEADS + h) * HDIM + dh) * SEQ + s0;
        *(v8h*)dst = pk;
      } else {
        const float scale = (mode == 0) ? 0.125f : 1.0f;   // 1/sqrt(64)
#pragma unroll
        for (int j = 0; j < 8; ++j)
          out[((size_t)(b * NHEADS + h) * SEQ + (s0 + j)) * HDIM + dh] =
              (_Float16)(acc[mi][ni][j] * scale);
      }
    }
}

// ---- Kernel 2: fused flash attention -----------------------------------
// Block = 8 waves = 8 query tiles of ONE (b,h). K/V tiles staged to LDS by
// async copies, double-buffered on ASYNCcnt, shared by all 8 waves.
__global__ __launch_bounds__(256) void mha_attn_kernel(
    const half_t* __restrict__ Qh, const half_t* __restrict__ Kh,
    const half_t* __restrict__ Vt, const unsigned char* __restrict__ mask,
    half_t* __restrict__ Oh) {
  __shared__ __attribute__((aligned(16))) half_t kbuf[2][32][64]; // 2x4KB
  __shared__ __attribute__((aligned(16))) half_t vbuf[2][64][32]; // 2x4KB
  __shared__ __attribute__((aligned(16))) half_t pbuf[8][16][40]; // 10KB

  const int tid    = threadIdx.x;
  const int lane   = tid & 31;
  const int wslot  = tid >> 5;
  const int QG     = SEQ / 16 / 8;                 // 16 q-groups
  const int qg     = blockIdx.x & (QG - 1);
  const int h      = (blockIdx.x / QG) & (NHEADS - 1);
  const int b      = blockIdx.x / (QG * NHEADS);
  const int qt     = qg * 8 + wslot;
  const int half16 = lane >> 4;
  const int g0     = half16 << 3;
  const int rloc   = lane & 15;

  const half_t* qbase =
      Qh + ((size_t)(b * NHEADS + h) * SEQ + qt * 16 + rloc) * HDIM;
  const v16h qa0 = pack_f16(qbase, g0);
  const v16h qa1 = pack_f16(qbase + 32, g0);

  const unsigned long long kbase =
      (unsigned long long)(uintptr_t)(Kh + (size_t)(b * NHEADS + h) * SEQ * HDIM);
  const unsigned long long vbase =
      (unsigned long long)(uintptr_t)(Vt + (size_t)(b * NHEADS + h) * HDIM * SEQ);
  const unsigned char* mbase = mask + (size_t)b * SEQ * SEQ;

  // per-thread async chunk assignment (16B each, 256 threads = full tile)
  const int krow = tid >> 3, kcoff = (tid & 7) * 8;      // kbuf[32][64]
  const int vrow = tid >> 2, vcoff = (tid & 3) * 8;      // vbuf[64][32]

  auto issue = [&](int kt, int bufsel) {
    async_ld16((unsigned)(uintptr_t)&kbuf[bufsel][krow][kcoff], kbase,
               (unsigned)(((kt + krow) * HDIM + kcoff) * sizeof(half_t)));
    async_ld16((unsigned)(uintptr_t)&vbuf[bufsel][vrow][vcoff], vbase,
               (unsigned)((vrow * SEQ + kt + vcoff) * sizeof(half_t)));
  };

  v8f oacc[4] = {{}, {}, {}, {}};
  float mj[8], lj[8];
#pragma unroll
  for (int j = 0; j < 8; ++j) { mj[j] = -1e30f; lj[j] = 0.f; }

  issue(0, 0);                                  // prologue: tile 0 -> buf 0
  const int NIT = SEQ / 32;                     // 64 iterations
  for (int it = 0; it < NIT; ++it) {
    const int cur = it & 1;
    const int kt  = it * 32;
    __syncthreads();                 // readers of buf cur^1 are done
    const int ktn = (it + 1 < NIT) ? kt + 32 : kt;   // dummy on last iter
    issue(ktn, cur ^ 1);
    wait_async_le2();                // this wave's buf[cur] loads landed
    __syncthreads();                 // buf[cur] valid block-wide

    // --- scores: two 16x16 D-tiles, K from LDS
    v8f s0 = {}, s1 = {};
    s0 = wmma_f16(qa0, pack_f16(&kbuf[cur][rloc][0],       g0), s0);
    s0 = wmma_f16(qa1, pack_f16(&kbuf[cur][rloc][32],      g0), s0);
    s1 = wmma_f16(qa0, pack_f16(&kbuf[cur][16 + rloc][0],  g0), s1);
    s1 = wmma_f16(qa1, pack_f16(&kbuf[cur][16 + rloc][32], g0), s1);

    // --- mask (True => -1e9)
#pragma unroll
    for (int j = 0; j < 8; ++j) {
      const int q = qt * 16 + j + (half16 << 3);
      const unsigned char* mrow = mbase + (size_t)q * SEQ + kt;
      if (mrow[rloc])      s0[j] = -1e9f;
      if (mrow[16 + rloc]) s1[j] = -1e9f;
    }

    // --- online softmax (row stats replicated per 16-lane half)
#pragma unroll
    for (int j = 0; j < 8; ++j) {
      float v = fmaxf(s0[j], s1[j]);
#pragma unroll
      for (int d = 8; d >= 1; d >>= 1) v = fmaxf(v, __shfl_xor(v, d, 32));
      const float nm    = fmaxf(mj[j], v);
      const float alpha = __expf(mj[j] - nm);
      const float p0 = __expf(s0[j] - nm);
      const float p1 = __expf(s1[j] - nm);
      s0[j] = p0; s1[j] = p1;
      float r = p0 + p1;
#pragma unroll
      for (int d = 8; d >= 1; d >>= 1) r += __shfl_xor(r, d, 32);
      lj[j] = lj[j] * alpha + r;
      mj[j] = nm;
      oacc[0][j] *= alpha; oacc[1][j] *= alpha;
      oacc[2][j] *= alpha; oacc[3][j] *= alpha;
    }

    // --- transpose P (D layout) -> A layout through LDS, as f16
    __syncthreads();
#pragma unroll
    for (int j = 0; j < 8; ++j) {
      const int m = j + (half16 << 3);
      pbuf[wslot][m][rloc]      = (_Float16)s0[j];
      pbuf[wslot][m][16 + rloc] = (_Float16)s1[j];
    }
    __syncthreads();
    const v16h pa = pack_f16(&pbuf[wslot][rloc][0], g0);

    // --- O += P @ V, V from LDS (rows = dh, contiguous along keys)
#pragma unroll
    for (int d = 0; d < 4; ++d)
      oacc[d] = wmma_f16(pa, pack_f16(&vbuf[cur][d * 16 + rloc][0], g0),
                         oacc[d]);
  }

  // --- normalize, store O as f16 [B,S,H*Dh]
  float inv[8];
#pragma unroll
  for (int j = 0; j < 8; ++j) inv[j] = (lj[j] > 0.f) ? (1.f / lj[j]) : 0.f;
#pragma unroll
  for (int d = 0; d < 4; ++d)
#pragma unroll
    for (int j = 0; j < 8; ++j) {
      const int srow = qt * 16 + j + (half16 << 3);
      const int col  = h * HDIM + d * 16 + rloc;
      Oh[(size_t)(b * SEQ + srow) * DMODEL + col] =
          (_Float16)(oacc[d][j] * inv[j]);
    }
}

// ---- Kernel 3: output projection  out = O @ w_o^T  (fp32 out) ----------
// Same 2x2 register blocking as the input projections.
__global__ __launch_bounds__(256) void mha_outproj_kernel(
    const half_t* __restrict__ Oh, const half_t* __restrict__ w,
    float* __restrict__ out) {
  const int lane   = threadIdx.x & 31;
  const int wave   = (blockIdx.x * blockDim.x + threadIdx.x) >> 5;
  const int TN2    = DMODEL / 32;
  const int tn2    = wave & (TN2 - 1);
  const int tm2    = wave / TN2;
  const int half16 = lane >> 4;
  const int g0     = half16 << 3;
  const int rloc   = lane & 15;

  const half_t* arow0 = Oh + (size_t)(tm2 * 32 + rloc) * DMODEL;
  const half_t* arow1 = arow0 + (size_t)16 * DMODEL;
  const half_t* wrow0 = w  + (size_t)(tn2 * 32 + rloc) * DMODEL;
  const half_t* wrow1 = wrow0 + (size_t)16 * DMODEL;

  v8f acc[2][2] = {{{}, {}}, {{}, {}}};
#pragma unroll 2
  for (int kk = 0; kk < DMODEL; kk += 32) {
    const v16h a0 = pack_f16(arow0 + kk, g0);
    const v16h a1 = pack_f16(arow1 + kk, g0);
    const v16h b0 = pack_f16(wrow0 + kk, g0);
    const v16h b1 = pack_f16(wrow1 + kk, g0);
    acc[0][0] = wmma_f16(a0, b0, acc[0][0]);
    acc[0][1] = wmma_f16(a0, b1, acc[0][1]);
    acc[1][0] = wmma_f16(a1, b0, acc[1][0]);
    acc[1][1] = wmma_f16(a1, b1, acc[1][1]);
  }

#pragma unroll
  for (int mi = 0; mi < 2; ++mi)
#pragma unroll
    for (int ni = 0; ni < 2; ++ni) {
      const int mrow0 = tm2 * 32 + mi * 16 + (half16 << 3);
      const int col   = tn2 * 32 + ni * 16 + rloc;
#pragma unroll
      for (int j = 0; j < 8; ++j)
        out[(size_t)(mrow0 + j) * DMODEL + col] = acc[mi][ni][j];
    }
}

// ---- host launch --------------------------------------------------------
extern "C" void kernel_launch(void* const* d_in, const int* in_sizes, int n_in,
                              void* d_out, int out_size, void* d_ws, size_t ws_size,
                              hipStream_t stream) {
  (void)in_sizes; (void)n_in; (void)out_size; (void)ws_size;
  const float* q   = (const float*)d_in[0];
  const float* k   = (const float*)d_in[1];
  const float* v   = (const float*)d_in[2];
  const unsigned char* mask = (const unsigned char*)d_in[3];  // jax bool, 1B
  const float* w_q = (const float*)d_in[4];
  const float* w_k = (const float*)d_in[5];
  const float* w_v = (const float*)d_in[6];
  const float* w_o = (const float*)d_in[7];
  float* out = (float*)d_out;

  const size_t E = (size_t)BATCH * SEQ * DMODEL;   // 4,194,304
  const size_t W = (size_t)DMODEL * DMODEL;        // 1,048,576
  half_t* p  = (half_t*)d_ws;                      // 64 MB total
  half_t* Qh = p;          half_t* Kh = p + E;
  half_t* Vt = p + 2 * E;  half_t* Oh = p + 3 * E;
  half_t* Xq = p + 4 * E;  half_t* Xk = p + 5 * E;  half_t* Xv = p + 6 * E;
  half_t* Wq = p + 7 * E;
  half_t* Wk = Wq + W;  half_t* Wv = Wq + 2 * W;  half_t* Wo = Wq + 3 * W;

  const dim3 blk(256);
  const dim3 gcvtX((unsigned)(E / 8 / 256));       // 2048
  const dim3 gcvtW((unsigned)(W / 8 / 256));       // 512
  const dim3 gproj((BATCH * SEQ / 32) * (DMODEL / 32) / 8);   // 512
  const dim3 gattn((BATCH * NHEADS * (SEQ / 16)) / 8);        // 512

  cvt_f16_kernel<<<gcvtX, blk, 0, stream>>>(q,   Xq, (int)(E / 8));
  cvt_f16_kernel<<<gcvtX, blk, 0, stream>>>(k,   Xk, (int)(E / 8));
  cvt_f16_kernel<<<gcvtX, blk, 0, stream>>>(v,   Xv, (int)(E / 8));
  cvt_f16_kernel<<<gcvtW, blk, 0, stream>>>(w_q, Wq, (int)(W / 8));
  cvt_f16_kernel<<<gcvtW, blk, 0, stream>>>(w_k, Wk, (int)(W / 8));
  cvt_f16_kernel<<<gcvtW, blk, 0, stream>>>(w_v, Wv, (int)(W / 8));
  cvt_f16_kernel<<<gcvtW, blk, 0, stream>>>(w_o, Wo, (int)(W / 8));

  mha_proj_kernel<<<gproj, blk, 0, stream>>>(Xq, Wq, Qh, 0);
  mha_proj_kernel<<<gproj, blk, 0, stream>>>(Xk, Wk, Kh, 1);
  mha_proj_kernel<<<gproj, blk, 0, stream>>>(Xv, Wv, Vt, 2);
  mha_attn_kernel<<<gattn, blk, 0, stream>>>(Qh, Kh, Vt, mask, Oh);
  mha_outproj_kernel<<<gproj, blk, 0, stream>>>(Oh, Wo, out);
}
